// TopKRouter_86157043958337
// MI455X (gfx1250) — compile-verified, matrix operand order checked by reading, not run on hardware
//
#include <hip/hip_runtime.h>

typedef __bf16 bf16_t;
typedef bf16_t v16bf __attribute__((ext_vector_type(16)));
typedef bf16_t v8bf  __attribute__((ext_vector_type(8)));
typedef float  v8f   __attribute__((ext_vector_type(8)));
typedef float  v4f   __attribute__((ext_vector_type(4)));

#define NUM_TOKENS  8192
#define DIM         4096
#define NUM_EXPERTS 64
#define TOP_K       8
#define KWAVES      8                      // waves per block = K-split factor
#define KSLICE      (DIM / KWAVES)         // 512 K elements per wave
#define KITER       (KSLICE / 32)          // 16 WMMA-K steps per wave

// ---------------------------------------------------------------------------
// Kernel 1: convert gate weights f32 -> bf16 (RNE) once into workspace.
// ---------------------------------------------------------------------------
__global__ __launch_bounds__(256)
void convert_gate_bf16(const float* __restrict__ gw,
                       unsigned short* __restrict__ out, int n4) {
  int i = blockIdx.x * blockDim.x + threadIdx.x;
  if (i >= n4) return;
  v4f f = *(const v4f*)(gw + (size_t)i * 4);
  unsigned int r[4];
#pragma unroll
  for (int j = 0; j < 4; ++j) {
    unsigned int u = __float_as_uint(f[j]);
    r[j] = (u + 0x7FFFu + ((u >> 16) & 1u)) >> 16;   // round-to-nearest-even
  }
  uint2 packed;
  packed.x = r[0] | (r[1] << 16);
  packed.y = r[2] | (r[3] << 16);
  *(uint2*)(out + (size_t)i * 4) = packed;
}

// -------- pipelined load helpers (distinct live values => no reg recycle) ---
struct AF { v4f x0, x1, x2, x3; };          // 16 f32 of the A slice
struct BV { v8bf lo[4], hi[4]; };           // 4 expert tiles, bf16

__device__ __forceinline__ AF load_a(const float* ap) {
  AF r;
  r.x0 = __builtin_nontemporal_load((const v4f*)(ap));       // K0..K0+3
  r.x1 = __builtin_nontemporal_load((const v4f*)(ap + 4));   // K0+4..K0+7
  r.x2 = __builtin_nontemporal_load((const v4f*)(ap + 16));  // K0+16..19
  r.x3 = __builtin_nontemporal_load((const v4f*)(ap + 20));  // K0+20..23
  return r;
}

__device__ __forceinline__ void load_b(const unsigned short* base, int off,
                                       BV& r) {
#pragma unroll
  for (int n = 0; n < 4; ++n) {
    const unsigned short* bp = base + (size_t)n * 16 * DIM + off;
    r.lo[n] = *(const v8bf*)(bp);         // 16B: K0..K0+7
    r.hi[n] = *(const v8bf*)(bp + 16);    // 16B: K0+16..K0+23
  }
}

__device__ __forceinline__ v16bf cvt_a(const AF& f) {
  v16bf a;
#pragma unroll
  for (int j = 0; j < 4; ++j) {
    a[j]      = (bf16_t)f.x0[j];
    a[4 + j]  = (bf16_t)f.x1[j];
    a[8 + j]  = (bf16_t)f.x2[j];
    a[12 + j] = (bf16_t)f.x3[j];
  }
  return a;
}

__device__ __forceinline__ v16bf cat(v8bf lo, v8bf hi) {
  return __builtin_shufflevector(lo, hi, 0, 1, 2, 3, 4, 5, 6, 7,
                                 8, 9, 10, 11, 12, 13, 14, 15);
}

// ---------------------------------------------------------------------------
// Kernel 2: fused gate GEMM (bf16 WMMA, f32 accum) + softmax + top-8.
// One block = one 16-token M-tile; 8 waves split K (512 each) for MLP.
// Partials combine via LDS ds_add_f32. 512 blocks x 256 threads = 4096 waves.
// ---------------------------------------------------------------------------
template <bool PRECONV>
__global__ __launch_bounds__(256)
void router_kernel(const float* __restrict__ hidden,
                   const float* __restrict__ gate_f32,
                   const unsigned short* __restrict__ gate_bf16,
                   const float* __restrict__ bias,
                   float* __restrict__ out_idx,
                   float* __restrict__ out_scr) {
  __shared__ float s_logits[16 * NUM_EXPERTS];   // 16 tokens x 64 experts
  __shared__ float s_keys[16 * NUM_EXPERTS];
  __shared__ float s_bias[NUM_EXPERTS];

  const int tid  = threadIdx.x;
  const int wave = tid >> 5;          // K-slice owned by this wave
  const int lane = tid & 31;

  if (tid < NUM_EXPERTS) s_bias[tid] = bias[tid];
#pragma unroll
  for (int j = 0; j < 4; ++j) s_logits[tid + 256 * j] = 0.f;
  __syncthreads();

  const int mTile = blockIdx.x;                 // 512 tiles
  const int row   = lane & 15;                  // M (A) / N (B) index
  const int khalf = lane >> 4;                  // which K-run set this lane owns
  const int token = mTile * 16 + row;
  const int kbase = wave * KSLICE;

  const float* aBase = hidden + (size_t)token * DIM + kbase + khalf * 8;

  v8f c0 = {}, c1 = {}, c2 = {}, c3 = {};

  if (PRECONV) {
    const unsigned short* bBase =
        gate_bf16 + (size_t)row * DIM + kbase + khalf * 8;

    // software pipeline: loads for kt+1 in flight while kt's WMMAs execute
    AF af = load_a(aBase);
    BV bv;
    load_b(bBase, 0, bv);

#pragma unroll 2
    for (int kt = 0; kt < KITER; ++kt) {
      const int ktn = (kt + 1 < KITER) ? kt + 1 : kt;   // clamped prefetch
      AF af_n = load_a(aBase + ktn * 32);
      BV bv_n;
      load_b(bBase, ktn * 32, bv_n);

      // fence: all 12 next-stage loads must issue BEFORE this point's math,
      // so the WMMA waits become partial (loadcnt ~12) instead of full drains
      __builtin_amdgcn_sched_barrier(0);

      v16bf a = cvt_a(af);
      c0 = __builtin_amdgcn_wmma_f32_16x16x32_bf16(
          false, a, false, cat(bv.lo[0], bv.hi[0]), (short)0, c0, false, false);
      c1 = __builtin_amdgcn_wmma_f32_16x16x32_bf16(
          false, a, false, cat(bv.lo[1], bv.hi[1]), (short)0, c1, false, false);
      c2 = __builtin_amdgcn_wmma_f32_16x16x32_bf16(
          false, a, false, cat(bv.lo[2], bv.hi[2]), (short)0, c2, false, false);
      c3 = __builtin_amdgcn_wmma_f32_16x16x32_bf16(
          false, a, false, cat(bv.lo[3], bv.hi[3]), (short)0, c3, false, false);

      __builtin_amdgcn_sched_barrier(0);

      af = af_n;
      bv = bv_n;
    }
  } else {
    // fallback: inline f32 -> bf16 conversion of gate weights
#pragma unroll 2
    for (int kt = 0; kt < KITER; ++kt) {
      v16bf a = cvt_a(load_a(aBase + kt * 32));
      v16bf b[4];
#pragma unroll
      for (int n = 0; n < 4; ++n) {
        const float* bp = gate_f32 +
            (size_t)(n * 16 + row) * DIM + kbase + kt * 32 + khalf * 8;
        AF bf = {*(const v4f*)(bp), *(const v4f*)(bp + 4),
                 *(const v4f*)(bp + 16), *(const v4f*)(bp + 20)};
        b[n] = cvt_a(bf);
      }
      c0 = __builtin_amdgcn_wmma_f32_16x16x32_bf16(false, a, false, b[0],
                                                   (short)0, c0, false, false);
      c1 = __builtin_amdgcn_wmma_f32_16x16x32_bf16(false, a, false, b[1],
                                                   (short)0, c1, false, false);
      c2 = __builtin_amdgcn_wmma_f32_16x16x32_bf16(false, a, false, b[2],
                                                   (short)0, c2, false, false);
      c3 = __builtin_amdgcn_wmma_f32_16x16x32_bf16(false, a, false, b[3],
                                                   (short)0, c3, false, false);
    }
  }

  // ---- combine K-slice partials: LDS float atomics (ds_add_f32) ----
  // C/D f32 16x16 layout: VGPR r holds M=r (lanes 0-15) and M=r+8 (lanes 16-31)
#pragma unroll
  for (int r = 0; r < 8; ++r) {
    const int tl = r + 8 * khalf;
    atomicAdd(&s_logits[tl * NUM_EXPERTS + 0  + row], c0[r]);
    atomicAdd(&s_logits[tl * NUM_EXPERTS + 16 + row], c1[r]);
    atomicAdd(&s_logits[tl * NUM_EXPERTS + 32 + row], c2[r]);
    atomicAdd(&s_logits[tl * NUM_EXPERTS + 48 + row], c3[r]);
  }
  __syncthreads();

  // ---- per-token softmax + biased top-8 (thread t owns token t) ----
  if (tid < 16) {
    float* L = &s_logits[tid * NUM_EXPERTS];
    float* K = &s_keys[tid * NUM_EXPERTS];

    float mx = -INFINITY;
#pragma unroll 4
    for (int e = 0; e < NUM_EXPERTS; ++e) mx = fmaxf(mx, L[e]);

    float sum = 0.f;
#pragma unroll 4
    for (int e = 0; e < NUM_EXPERTS; ++e) {
      float ex = __expf(L[e] - mx);
      L[e] = ex;            // overwrite logits with unnormalized probs
      sum += ex;
    }
    const float inv = 1.f / sum;

#pragma unroll 4
    for (int e = 0; e < NUM_EXPERTS; ++e)
      K[e] = L[e] * inv + s_bias[e];   // selection key = prob + bias

    float sel_s[TOP_K];
    int   sel_i[TOP_K];
    float ssum = 0.f;
    for (int k = 0; k < TOP_K; ++k) {
      float best = -INFINITY;
      int   bi   = 0;
#pragma unroll 4
      for (int e = 0; e < NUM_EXPERTS; ++e) {
        float v = K[e];
        if (v > best) { best = v; bi = e; }   // strict > => lowest index wins
      }
      K[bi] = -INFINITY;
      float sc = L[bi] * inv;
      sel_i[k] = bi;
      sel_s[k] = sc;
      ssum += sc;
    }
    const float rn = 1.f / (ssum + 1e-20f);

    const int tok = mTile * 16 + tid;
#pragma unroll
    for (int k = 0; k < TOP_K; ++k) {
      out_idx[tok * TOP_K + k] = (float)sel_i[k];
      out_scr[tok * TOP_K + k] = sel_s[k] * rn;
    }
  }
}

// ---------------------------------------------------------------------------
extern "C" void kernel_launch(void* const* d_in, const int* in_sizes, int n_in,
                              void* d_out, int out_size, void* d_ws,
                              size_t ws_size, hipStream_t stream) {
  (void)in_sizes; (void)n_in; (void)out_size;
  const float* hidden = (const float*)d_in[0];
  const float* gate   = (const float*)d_in[1];
  const float* bias   = (const float*)d_in[2];

  float* out_idx = (float*)d_out;                              // [T, 8]
  float* out_scr = (float*)d_out + (size_t)NUM_TOKENS * TOP_K; // [T, 8]

  const size_t need = (size_t)NUM_EXPERTS * DIM * sizeof(unsigned short);
  if (d_ws && ws_size >= need) {
    unsigned short* gbf = (unsigned short*)d_ws;
    const int n4 = NUM_EXPERTS * DIM / 4;
    convert_gate_bf16<<<(n4 + 255) / 256, 256, 0, stream>>>(gate, gbf, n4);
    router_kernel<true><<<NUM_TOKENS / 16, 256, 0, stream>>>(
        hidden, gate, gbf, bias, out_idx, out_scr);
  } else {
    router_kernel<false><<<NUM_TOKENS / 16, 256, 0, stream>>>(
        hidden, gate, nullptr, bias, out_idx, out_scr);
  }
}
